// AdaptiveFusionDecoder_64166811402964
// MI455X (gfx1250) — matrix operand order repaired and checked by reading, not compiled, not run on hardware
//
#include <hip/hip_runtime.h>
#include <hip/hip_bf16.h>
#include <math.h>

// AdaptiveFusionDecoder for MI455X (gfx1250, wave32, WMMA bf16).
// Strategy: sequential GRU recurrence in bf16 WMMA (weights L2-resident),
// vocab projection hoisted into one big [2048,1024]x[1024,32000] WMMA GEMM
// with async global->LDS double-buffered B-tile staging (ASYNCcnt path).

#define B_  32
#define S_  64
#define E_  512
#define H_  1024
#define V_  32000
#define K2E (2 * E_)   // 1024 (GRU input width)
#define H3  (3 * H_)   // 3072

typedef unsigned short bf16_t;
typedef __attribute__((ext_vector_type(16))) __bf16 v16bf;
typedef __attribute__((ext_vector_type(8)))  float  v8f;

union V16 { uint4 q[2]; v16bf v; };

__device__ __forceinline__ bf16_t f2bf(float f) {
  union { float f; unsigned u; } c; c.f = f;
  unsigned u = c.u;
  u += 0x7fffu + ((u >> 16) & 1u);   // round-to-nearest-even
  return (bf16_t)(u >> 16);
}
__device__ __forceinline__ float sigmoidf_(float x) { return 1.0f / (1.0f + __expf(-x)); }

// ---- WMMA operand loads (wave32; this code's contiguous k order, not the ISA table order) ----------
// A-matrix 16x32 bf16: lane l(<16)=row l holds K k0+0..15 (v0-7);
// lane l+16 = row l holds K k0+16..31.
__device__ __forceinline__ v16bf load_A(const bf16_t* __restrict__ tile,
                                        int ldk, int k0, int lane) {
  int row  = lane & 15;
  int koff = k0 + ((lane >> 4) ? 16 : 0);
  const bf16_t* p = tile + (size_t)row * ldk + koff;
  V16 r;
  r.q[0] = *(const uint4*)(p);        // K koff..koff+7
  r.q[1] = *(const uint4*)(p + 8);    // K koff+8..koff+15
  return r.v;
}
// B-matrix 32x16 bf16 where B(k,j) = W[nbase+j][k] (W row-major, ld=ldk):
// lane l holds column j=l&15, K = k0 + (l>=16?16:0) + 0..15 (32B contiguous).
__device__ __forceinline__ v16bf load_B(const bf16_t* __restrict__ wrow0,
                                        int ldk, int k0, int lane) {
  int j    = lane & 15;
  int koff = k0 + ((lane >> 4) ? 16 : 0);
  const bf16_t* p = wrow0 + (size_t)j * ldk + koff;
  V16 r;
  r.q[0] = *(const uint4*)(p);
  r.q[1] = *(const uint4*)(p + 8);
  return r.v;
}
// Same B layout but sourced from an LDS tile stored as [col][32 k] bf16.
__device__ __forceinline__ v16bf load_B_lds(const bf16_t* tile32, int lane) {
  int j    = lane & 15;
  int koff = (lane >> 4) ? 16 : 0;
  const bf16_t* p = tile32 + j * 32 + koff;
  V16 r;
  r.q[0] = *(const uint4*)(p);
  r.q[1] = *(const uint4*)(p + 8);
  return r.v;
}
__device__ __forceinline__ v8f wmma_bf16(v16bf a, v16bf b, v8f c) {
  return __builtin_amdgcn_wmma_f32_16x16x32_bf16(false, a, false, b, (short)0, c,
                                                 false, false);
}

// ---- CDNA5 async global->LDS copy (ASYNCcnt; CDNA5 ISA).
// VDST holds the per-lane LDS byte address (flat LDS ptr truncated to 32 bits:
// aperture mapping uses addr[31:0]); VADDR holds the 64-bit global address.
__device__ __forceinline__ void async_copy_b128(void* lds_dst, const void* gsrc) {
  unsigned lds_off = (unsigned)(size_t)lds_dst;
  asm volatile("global_load_async_to_lds_b128 %0, %1, off"
               :: "v"(lds_off), "v"(gsrc) : "memory");
}
__device__ __forceinline__ void wait_asynccnt0() {
  asm volatile("s_wait_asynccnt 0x0" ::: "memory");
}

// ---- f32 -> bf16 weight conversion (vectorized) ---------------------------
__global__ void k_f32_to_bf16_v4(const float4* __restrict__ src,
                                 ushort4* __restrict__ dst, int n4) {
  int i = blockIdx.x * blockDim.x + threadIdx.x;
  if (i < n4) {
    float4 f = src[i];
    ushort4 o;
    o.x = f2bf(f.x); o.y = f2bf(f.y); o.z = f2bf(f.z); o.w = f2bf(f.w);
    dst[i] = o;
  }
}

// ---- token embedding gather -> bf16 [B*S][E] ------------------------------
__global__ void k_gather_tokens(const int* __restrict__ tokens,
                                const float* __restrict__ tok_embed,
                                bf16_t* __restrict__ tok_bf) {
  long long i = (long long)blockIdx.x * blockDim.x + threadIdx.x;
  if (i >= (long long)B_ * S_ * E_) return;
  int e = (int)(i % E_);
  long long bs = i / E_;
  int tk = tokens[bs];
  tok_bf[i] = f2bf(tok_embed[(long long)tk * E_ + e]);
}

// ---- h0 = tanh(image_emb @ init_h_W^T + b) --------------------------------
__global__ void k_init_h(const float* __restrict__ img,
                         const float* __restrict__ W,
                         const float* __restrict__ b,
                         float* __restrict__ hf, bf16_t* __restrict__ hbf) {
  int i = blockIdx.x * blockDim.x + threadIdx.x;
  if (i >= B_ * H_) return;
  int bb = i / H_, j = i % H_;
  const float4* x = (const float4*)(img + (size_t)bb * E_);
  const float4* w = (const float4*)(W + (size_t)j * E_);
  float s = 0.f;
  for (int k = 0; k < E_ / 4; ++k) {
    float4 a = x[k], c = w[k];
    s += a.x * c.x + a.y * c.y + a.z * c.z + a.w * c.w;
  }
  float v = tanhf(s + b[j]);
  hf[i] = v;
  hbf[i] = f2bf(v);
}

// ---- per-step gate: alpha, fused, assemble gru_in (bf16 [32][1024]) -------
__global__ void k_gate(int t, const float* __restrict__ hf,
                       const float* __restrict__ img,
                       const float* __restrict__ ret,
                       const float* __restrict__ gate_W,
                       const float* __restrict__ gate_b,
                       const bf16_t* __restrict__ tok_bf,
                       bf16_t* __restrict__ gruin) {
  __shared__ float part[256];
  __shared__ float alpha[B_];
  int tid = threadIdx.x;
  int b = tid >> 3, c = tid & 7;
  float s = 0.f;
  for (int idx = c * 256; idx < (c + 1) * 256; ++idx) {
    float wi = gate_W[idx], xi;
    if (idx < H_)            xi = hf[(size_t)b * H_ + idx];
    else if (idx < H_ + E_)  xi = img[(size_t)b * E_ + idx - H_];
    else                     xi = ret[(size_t)b * E_ + idx - H_ - E_];
    s += xi * wi;
  }
  part[tid] = s;
  __syncthreads();
  if (tid < B_) {
    float acc = 0.f;
    for (int k = 0; k < 8; ++k) acc += part[tid * 8 + k];
    alpha[tid] = sigmoidf_(acc + gate_b[0]);
  }
  __syncthreads();
  for (int i = tid; i < B_ * E_; i += 256) {
    int bb = i / E_, e = i % E_;
    float a = alpha[bb];
    float f = a * img[(size_t)bb * E_ + e] + (1.f - a) * ret[(size_t)bb * E_ + e];
    gruin[(size_t)bb * K2E + E_ + e] = f2bf(f);                           // fused
    gruin[(size_t)bb * K2E + e] = tok_bf[((size_t)bb * S_ + t) * E_ + e]; // x_t
  }
}

// ---- one GRU step: G = gru_in@Wih^T + h@Whh^T, fused nonlinearity ---------
// 128 waves; wave = one (16 rows x 16 cols) r/z/n triple. h double-buffered.
__global__ void __launch_bounds__(256) k_gru_step(
    int t,
    const bf16_t* __restrict__ gruin,  // [32][1024]
    const bf16_t* __restrict__ hbf,    // [32][1024] current (read)
    const float*  __restrict__ hf,     // [32][1024] current (read)
    const bf16_t* __restrict__ Wih,    // [3072][1024] bf16
    const bf16_t* __restrict__ Whh,    // [3072][1024] bf16
    const float* __restrict__ bih, const float* __restrict__ bhh,
    float* __restrict__ hf_out, bf16_t* __restrict__ hbf_out,
    bf16_t* __restrict__ hseq)         // [B*S][H], row = b*S + t
{
  int wave = threadIdx.x >> 5;
  int lane = threadIdx.x & 31;
  int unit = blockIdx.x * 8 + wave;    // 0..127
  int mt = unit >> 6;                  // 0..1   (16-row batch tile)
  int nb = (unit & 63) * 16;           // 0..1008 (hidden column tile)

  const bf16_t* Ag_base = gruin + (size_t)mt * 16 * K2E;
  const bf16_t* Ah_base = hbf   + (size_t)mt * 16 * H_;
  const bf16_t* Br_ih = Wih + (size_t)(0 * H_ + nb) * K2E;
  const bf16_t* Bz_ih = Wih + (size_t)(1 * H_ + nb) * K2E;
  const bf16_t* Bn_ih = Wih + (size_t)(2 * H_ + nb) * K2E;
  const bf16_t* Br_hh = Whh + (size_t)(0 * H_ + nb) * H_;
  const bf16_t* Bz_hh = Whh + (size_t)(1 * H_ + nb) * H_;
  const bf16_t* Bn_hh = Whh + (size_t)(2 * H_ + nb) * H_;

  v8f ar = {}, az = {}, ain = {}, ahn = {};
  for (int k0 = 0; k0 < 1024; k0 += 32) {
    v16bf Ag = load_A(Ag_base, K2E, k0, lane);
    v16bf Ah = load_A(Ah_base, H_,  k0, lane);
    ar  = wmma_bf16(Ag, load_B(Br_ih, K2E, k0, lane), ar);   // i_r
    ar  = wmma_bf16(Ah, load_B(Br_hh, H_,  k0, lane), ar);   // +h_r
    az  = wmma_bf16(Ag, load_B(Bz_ih, K2E, k0, lane), az);   // i_z
    az  = wmma_bf16(Ah, load_B(Bz_hh, H_,  k0, lane), az);   // +h_z
    ain = wmma_bf16(Ag, load_B(Bn_ih, K2E, k0, lane), ain);  // i_n (separate!)
    ahn = wmma_bf16(Ah, load_B(Bn_hh, H_,  k0, lane), ahn);  // h_n (separate!)
  }

  int mloc = (lane >> 4) * 8;
  int col  = nb + (lane & 15);
  float br  = bih[col] + bhh[col];
  float bz  = bih[H_ + col] + bhh[H_ + col];
  float bin = bih[2 * H_ + col];
  float bhn = bhh[2 * H_ + col];
#pragma unroll
  for (int g = 0; g < 8; ++g) {
    int b = mt * 16 + mloc + g;
    float r  = sigmoidf_(ar[g] + br);
    float z  = sigmoidf_(az[g] + bz);
    float nn = tanhf(ain[g] + bin + r * (ahn[g] + bhn));
    float hp = hf[(size_t)b * H_ + col];
    float hn = (1.f - z) * nn + z * hp;
    hf_out[(size_t)b * H_ + col] = hn;
    bf16_t hb = f2bf(hn);
    hbf_out[(size_t)b * H_ + col] = hb;
    hseq[((size_t)b * S_ + t) * H_ + col] = hb;
  }
}

// ---- final projection: [2048,1024]bf16 x [1024,32000]bf16 -> f32 logits ---
// WG = 256 rows x 64 cols. The shared 64x32k B tile (4KB) is staged into LDS
// once per k-step via async global->LDS b128 (one op per thread, per-wave
// s_wait_asynccnt + workgroup barrier), double-buffered. Each of the 8 waves
// computes a 32x64 C block (2 M-tiles x 4 N-tiles) from the LDS tile, so B is
// read from L2 once per 256-row block instead of once per 32-row wave (8x cut).
__global__ void __launch_bounds__(256) k_logits(
    const bf16_t* __restrict__ A,   // hseq [2048][1024]
    const bf16_t* __restrict__ Wo,  // [32000][1024]
    const float* __restrict__ bo,   // [32000]
    float* __restrict__ out)        // [2048][32000]
{
  __shared__ bf16_t smB[2][64 * 32];   // [buf][col][k] bf16, 2 x 4KB

  int tid  = threadIdx.x;
  int wave = tid >> 5, lane = tid & 31;
  int mb = blockIdx.x & 7;             // 0..7   (256-row block)
  int nt = blockIdx.x >> 3;            // 0..499 (64-col block)
  int mbase = mb * 256 + wave * 32;
  int nbase = nt * 64;
  const bf16_t* A0 = A + (size_t)mbase * H_;
  const bf16_t* A1 = A + (size_t)(mbase + 16) * H_;

  // Per-thread staging slice: 16B of Wo[(nbase+col), k0 + kc*8 ..]
  int scol = tid >> 2, skc = (tid & 3) * 8;
  const bf16_t* sgsrc = Wo + (size_t)(nbase + scol) * H_ + skc;
  bf16_t* sdst0 = &smB[0][scol * 32 + skc];
  bf16_t* sdst1 = &smB[1][scol * 32 + skc];

  // Prologue: stage k-step 0 into buffer 0.
  async_copy_b128(sdst0, sgsrc);
  wait_asynccnt0();
  __syncthreads();

  v8f acc[2][4] = {};
  for (int i = 0; i < 32; ++i) {
    int k0 = i * 32;
    int cur = i & 1;
    if (i + 1 < 32)   // prefetch next B tile into the other buffer
      async_copy_b128(cur ? sdst0 : sdst1, sgsrc + (k0 + 32));

    v16bf a0 = load_A(A0, H_, k0, lane);
    v16bf a1 = load_A(A1, H_, k0, lane);
#pragma unroll
    for (int sn = 0; sn < 4; ++sn) {
      v16bf b = load_B_lds(&smB[cur][sn * 16 * 32], lane);
      acc[0][sn] = wmma_bf16(a0, b, acc[0][sn]);
      acc[1][sn] = wmma_bf16(a1, b, acc[1][sn]);
    }
    wait_asynccnt0();   // own wave's prefetch slice landed
    __syncthreads();    // all waves: compute done + full tile visible
  }

  int mloc = (lane >> 4) * 8, cloc = lane & 15;
#pragma unroll
  for (int sn = 0; sn < 4; ++sn) {
    int col = nbase + sn * 16 + cloc;
    float bias = bo[col];
#pragma unroll
    for (int sm = 0; sm < 2; ++sm) {
#pragma unroll
      for (int g = 0; g < 8; ++g) {
        int row = mbase + sm * 16 + mloc + g;
        out[(size_t)row * V_ + col] = acc[sm][sn][g] + bias;
      }
    }
  }
}

extern "C" void kernel_launch(void* const* d_in, const int* in_sizes, int n_in,
                              void* d_out, int out_size, void* d_ws, size_t ws_size,
                              hipStream_t stream) {
  const float* image_emb = (const float*)d_in[0];
  const float* retrieved = (const float*)d_in[1];
  const int*   tokens    = (const int*)d_in[2];
  const float* tok_embed = (const float*)d_in[3];
  const float* init_h_W  = (const float*)d_in[4];
  const float* init_h_b  = (const float*)d_in[5];
  const float* gru_W_ih  = (const float*)d_in[6];
  const float* gru_W_hh  = (const float*)d_in[7];
  const float* gru_b_ih  = (const float*)d_in[8];
  const float* gru_b_hh  = (const float*)d_in[9];
  const float* gate_W    = (const float*)d_in[10];
  const float* gate_b    = (const float*)d_in[11];
  const float* out_W     = (const float*)d_in[12];
  const float* out_b     = (const float*)d_in[13];
  float* out = (float*)d_out;

  // Workspace carve-up (~85 MB; 256B-aligned chunks).
  char* ws = (char*)d_ws;
  size_t off = 0;
  auto alloc = [&](size_t bytes) -> void* {
    void* p = ws + off;
    off = (off + bytes + 255) & ~(size_t)255;
    return p;
  };
  bf16_t* Wih_bf = (bf16_t*)alloc((size_t)H3 * K2E * 2);
  bf16_t* Whh_bf = (bf16_t*)alloc((size_t)H3 * H_ * 2);
  bf16_t* Wo_bf  = (bf16_t*)alloc((size_t)V_ * H_ * 2);
  bf16_t* tok_bf = (bf16_t*)alloc((size_t)B_ * S_ * E_ * 2);
  bf16_t* gruin  = (bf16_t*)alloc((size_t)B_ * K2E * 2);
  float*  hf0    = (float*)alloc((size_t)B_ * H_ * 4);
  float*  hf1    = (float*)alloc((size_t)B_ * H_ * 4);
  bf16_t* hb0    = (bf16_t*)alloc((size_t)B_ * H_ * 2);
  bf16_t* hb1    = (bf16_t*)alloc((size_t)B_ * H_ * 2);
  bf16_t* hseq   = (bf16_t*)alloc((size_t)B_ * S_ * H_ * 2);
  (void)ws_size; (void)in_sizes; (void)n_in; (void)out_size;

  // One-time bf16 weight conversions (L2/HBM streaming, ~10us total).
  {
    int n4;
    n4 = H3 * K2E / 4;
    k_f32_to_bf16_v4<<<(n4 + 255) / 256, 256, 0, stream>>>(
        (const float4*)gru_W_ih, (ushort4*)Wih_bf, n4);
    n4 = H3 * H_ / 4;
    k_f32_to_bf16_v4<<<(n4 + 255) / 256, 256, 0, stream>>>(
        (const float4*)gru_W_hh, (ushort4*)Whh_bf, n4);
    n4 = V_ * H_ / 4;
    k_f32_to_bf16_v4<<<(n4 + 255) / 256, 256, 0, stream>>>(
        (const float4*)out_W, (ushort4*)Wo_bf, n4);
  }
  {
    long long n = (long long)B_ * S_ * E_;
    k_gather_tokens<<<(int)((n + 255) / 256), 256, 0, stream>>>(tokens, tok_embed,
                                                                tok_bf);
  }
  k_init_h<<<(B_ * H_ + 255) / 256, 256, 0, stream>>>(image_emb, init_h_W,
                                                      init_h_b, hf0, hb0);

  float*  hf[2]  = {hf0, hf1};
  bf16_t* hbf[2] = {hb0, hb1};
  int cur = 0;
  for (int t = 0; t < S_; ++t) {
    k_gate<<<1, 256, 0, stream>>>(t, hf[cur], image_emb, retrieved, gate_W,
                                  gate_b, tok_bf, gruin);
    k_gru_step<<<16, 256, 0, stream>>>(t, gruin, hbf[cur], hf[cur], Wih_bf,
                                       Whh_bf, gru_b_ih, gru_b_hh, hf[1 - cur],
                                       hbf[1 - cur], hseq);
    cur = 1 - cur;
  }
  // Hoisted vocab projection: 134 GFLOP bf16 WMMA, async-LDS staged B.
  k_logits<<<4000, 256, 0, stream>>>(hseq, Wo_bf, out_b, out);
}
